// FilterMLPBlock_28793460753007
// MI455X (gfx1250) — matrix operand order repaired and checked by reading, not compiled, not run on hardware
//
#include <hip/hip_runtime.h>
#include <hip/hip_bf16.h>
#include <math.h>

// CDNA5 / gfx1250, wave32.
typedef __attribute__((ext_vector_type(16))) __bf16 v16bf;
typedef __attribute__((ext_vector_type(8)))  __bf16 v8bf;
typedef __attribute__((ext_vector_type(8)))  float  v8f;

#define Bsz 4096
#define Ssz 64
#define Dsz 256
#define DC  32     // d-channels per workgroup in GEMM kernel
#define TB  16     // batch rows per workgroup (one WMMA M tile)
#define APITCH 72  // LDS row pitch in bf16 units (144B, 16B aligned for ds_load_b128)

// ---------------------------------------------------------------------------
// Kernel 1: build per-d circulant kernels k_d[r] and pre-swizzle them into
// WMMA B-matrix fragments (bf16 hi/lo split) in d_ws.
// ws layout (bf16): [d][prec(hi=0,lo=1)][kc(0,1)][st(0..3)][lane(0..31)][e(0..15)]
//   B tile for (kc,st) is Mt[t0+K][s0+N], Mt[t][s] = k_d[(s-t)&63],
//   lane L: N = L%16, K = (L/16)*8 + (e&7) + (e>>3)*16   (16-bit B layout, per ISA 05_wmma)
// ---------------------------------------------------------------------------
__global__ void k_build_frags(const float* __restrict__ cw,
                              unsigned short* __restrict__ wsB_u16)
{
    __bf16* wsB = (__bf16*)wsB_u16;
    __shared__ float kf[64];
    const int d   = blockIdx.x;
    const int tid = threadIdx.x;            // 64 threads
    {
        const int r = tid;
        const float wr0  = cw[ 0 * Dsz * 2 + d * 2 + 0];
        const float wr32 = cw[32 * Dsz * 2 + d * 2 + 0];
        float s = wr0 + ((r & 1) ? -wr32 : wr32);
        const float c0 = 6.28318530717958647692f / 64.0f;
        for (int f = 1; f <= 31; ++f) {
            const float wr  = cw[f * Dsz * 2 + d * 2 + 0];
            const float wi  = cw[f * Dsz * 2 + d * 2 + 1];
            const float ang = c0 * (float)(f * r);
            s += 2.0f * (wr * __cosf(ang) - wi * __sinf(ang));
        }
        kf[r] = s * (1.0f / 64.0f);
    }
    __syncthreads();
    for (int i = 0; i < 128; ++i) {
        const int idx  = tid + i * 64;      // 0..8191
        const int e    =  idx        & 15;
        const int lane = (idx >> 4)  & 31;
        const int st   = (idx >> 9)  & 3;
        const int kc   = (idx >> 11) & 1;
        const int prec = (idx >> 12) & 1;
        const int K = ((lane >> 4) << 3) + (e & 7) + ((e >> 3) << 4);
        const int t = kc * 32 + K;
        const int s = st * 16 + (lane & 15);
        const float kv = kf[(s - t) & 63];
        const __bf16 hi = (__bf16)kv;
        const __bf16 v  = prec ? (__bf16)(kv - (float)hi) : hi;
        wsB[(size_t)d * 8192 + idx] = v;
    }
}

// ---------------------------------------------------------------------------
// Kernel 2: batched circulant GEMM, bf16 WMMA with hi/lo split (3 products).
// Grid: (B/16, D/32); block 256 = 8 waves; each wave owns 4 d-channels.
// Writes h = y + x TRANSPOSED to ht[B][D][S] (d_out used as staging).
// ---------------------------------------------------------------------------
__global__ void __launch_bounds__(256)
k_gemm(const float* __restrict__ x,
       const unsigned short* __restrict__ wsB_u16,
       float* __restrict__ ht)
{
    extern __shared__ __bf16 smem[];
    __bf16* Ahi = smem;                       // [DC][TB][APITCH]
    __bf16* Alo = smem + DC * TB * APITCH;

    const int b0  = blockIdx.x * TB;
    const int d0  = blockIdx.y * DC;
    const int tid = threadIdx.x;

    // ---- stage x tile -> LDS as bf16 hi/lo, layout [dd][bb][t] (t-contig) ----
    for (int i = 0; i < 32; ++i) {
        const int idx = tid + i * 256;        // 8192 float4 loads total
        const int q   = idx & 7;              // which float4 in 32-d row
        const int row = idx >> 3;             // bb*64 + t
        const int bb  = row >> 6;
        const int t   = row & 63;
        const float4 v =
            *(const float4*)(x + (size_t)(b0 + bb) * Ssz * Dsz + t * Dsz + d0 + q * 4);
        const float vv[4] = {v.x, v.y, v.z, v.w};
        #pragma unroll
        for (int j = 0; j < 4; ++j) {
            const int dd = q * 4 + j;
            const __bf16 hi = (__bf16)vv[j];
            const __bf16 lo = (__bf16)(vv[j] - (float)hi);
            const int a = (dd * TB + bb) * APITCH + t;
            Ahi[a] = hi;
            Alo[a] = lo;
        }
    }
    __syncthreads();

    const int w    = tid >> 5;
    const int lane = tid & 31;
    const int ln   = lane & 15;               // A: M row / B: N col / C: N col
    const int lh   = lane >> 4;

    for (int dd4 = 0; dd4 < 4; ++dd4) {
        const int dd = w * 4 + dd4;
        const int d  = d0 + dd;

        // A fragments: lane L holds M=L%16; elems j<8 -> K=lh*8+j, j>=8 -> K=16+lh*8+(j-8)
        v16bf ahiF[2], aloF[2];
        #pragma unroll
        for (int kc = 0; kc < 2; ++kc) {
            const int base = (dd * TB + ln) * APITCH + kc * 32 + lh * 8;
            const v8bf h0 = *(const v8bf*)(Ahi + base);
            const v8bf h1 = *(const v8bf*)(Ahi + base + 16);
            const v8bf l0 = *(const v8bf*)(Alo + base);
            const v8bf l1 = *(const v8bf*)(Alo + base + 16);
            ahiF[kc] = __builtin_shufflevector(h0, h1,
                         0,1,2,3,4,5,6,7,8,9,10,11,12,13,14,15);
            aloF[kc] = __builtin_shufflevector(l0, l1,
                         0,1,2,3,4,5,6,7,8,9,10,11,12,13,14,15);
        }

        const v16bf* bfD = (const v16bf*)wsB_u16 + (size_t)d * 512;

        #pragma unroll
        for (int st = 0; st < 4; ++st) {
            v8f acc = {};
            #pragma unroll
            for (int kc = 0; kc < 2; ++kc) {
                const v16bf bh = bfD[((0 * 2 + kc) * 4 + st) * 32 + lane];
                const v16bf bl = bfD[((1 * 2 + kc) * 4 + st) * 32 + lane];
                // y ≈ A_lo*B_hi + A_hi*B_lo + A_hi*B_hi  (small terms first)
                acc = __builtin_amdgcn_wmma_f32_16x16x32_bf16(
                        false, aloF[kc], false, bh, (short)0, acc, false, false);
                acc = __builtin_amdgcn_wmma_f32_16x16x32_bf16(
                        false, ahiF[kc], false, bl, (short)0, acc, false, false);
                acc = __builtin_amdgcn_wmma_f32_16x16x32_bf16(
                        false, ahiF[kc], false, bh, (short)0, acc, false, false);
            }
            // C layout: VGPR r, lane L: M=(L/16)*8+r, N=L%16
            const int s = st * 16 + ln;
            #pragma unroll
            for (int r = 0; r < 8; ++r) {
                const int m = lh * 8 + r;
                const int a = (dd * TB + m) * APITCH + s;
                const float xv = (float)Ahi[a] + (float)Alo[a];   // residual (≈fp32)
                ht[((size_t)(b0 + m) * Dsz + d) * Ssz + s] = acc[r] + xv;
            }
        }
    }
}

// ---------------------------------------------------------------------------
// Kernel 3: LayerNorm over D, reading ht[B][D][S], writing out[B][S][D]
// in place over d_out. One block per b.
// ---------------------------------------------------------------------------
__global__ void __launch_bounds__(256)
k_layernorm(const float* __restrict__ gamma, const float* __restrict__ beta,
            float* __restrict__ out)
{
    extern __shared__ float sm[];
    float* hl  = sm;              // [256][65] padded
    float* ps  = hl + 256 * 65;   // [4][64] partial sums
    float* pq  = ps + 256;        // [4][64] partial sumsq
    float* mus = pq + 256;        // [64]
    float* rss = mus + 64;        // [64]

    const int b   = blockIdx.x;
    const int tid = threadIdx.x;
    const float* ht = out;        // same buffer, this block's slice only

    for (int i = 0; i < 64; ++i) {
        const int idx = tid + i * 256;         // 0..16383, s-contiguous
        const int d = idx >> 6, s = idx & 63;
        hl[d * 65 + s] = ht[(size_t)b * 16384 + idx];
    }
    __syncthreads();

    {
        const int s = tid & 63, q = tid >> 6;
        float sum = 0.f, sq = 0.f;
        for (int d = q * 64; d < q * 64 + 64; ++d) {
            const float v = hl[d * 65 + s];
            sum += v; sq += v * v;
        }
        ps[q * 64 + s] = sum;
        pq[q * 64 + s] = sq;
    }
    __syncthreads();
    if (tid < 64) {
        const int s = tid;
        const float sum = ps[s] + ps[64 + s] + ps[128 + s] + ps[192 + s];
        const float sq  = pq[s] + pq[64 + s] + pq[128 + s] + pq[192 + s];
        const float mu  = sum * (1.0f / 256.0f);
        const float var = sq * (1.0f / 256.0f) - mu * mu;
        mus[s] = mu;
        rss[s] = rsqrtf(fmaxf(var, 0.0f) + 1e-12f);
    }
    __syncthreads();

    for (int i = 0; i < 64; ++i) {
        const int idx = tid + i * 256;         // out[b][s][d], d-contiguous
        const int d = idx & 255, s = idx >> 8;
        out[(size_t)b * 16384 + idx] =
            (hl[d * 65 + s] - mus[s]) * rss[s] * gamma[d] + beta[d];
    }
}

// ---------------------------------------------------------------------------
extern "C" void kernel_launch(void* const* d_in, const int* in_sizes, int n_in,
                              void* d_out, int out_size, void* d_ws, size_t ws_size,
                              hipStream_t stream)
{
    (void)in_sizes; (void)n_in; (void)out_size; (void)ws_size;
    const float* x     = (const float*)d_in[0];
    const float* cw    = (const float*)d_in[1];   // [1,33,256,2]
    const float* gamma = (const float*)d_in[2];
    const float* beta  = (const float*)d_in[3];
    float* out = (float*)d_out;
    unsigned short* wsB = (unsigned short*)d_ws;  // 4 MiB of bf16 B-fragments

    // 1) circulant kernels -> WMMA-swizzled bf16 hi/lo fragments (L2-resident)
    k_build_frags<<<256, 64, 0, stream>>>(cw, wsB);

    // 2) batched GEMM via v_wmma_f32_16x16x32_bf16, residual add, transpose-store
    dim3 g2(Bsz / TB, Dsz / DC);
    const size_t lds2 = (size_t)2 * DC * TB * APITCH * sizeof(unsigned short); // 147456 B
    k_gemm<<<g2, 256, lds2, stream>>>(x, wsB, out);

    // 3) LayerNorm over D, in place on d_out
    const size_t lds3 = (size_t)(256 * 65 + 256 + 256 + 64 + 64) * sizeof(float); // 69120 B
    k_layernorm<<<Bsz, 256, lds3, stream>>>(gamma, beta, out);
}
